// MultiHeadAttention_1632087572645
// MI455X (gfx1250) — matrix operand order repaired and checked by reading, not compiled, not run on hardware
//
#include <hip/hip_runtime.h>
#include <hip/hip_bf16.h>

typedef __attribute__((ext_vector_type(16))) _Float16 v16h;
typedef __attribute__((ext_vector_type(8)))  _Float16 v8h;
typedef __attribute__((ext_vector_type(8)))  float    v8f;

static constexpr int BB = 4;
static constexpr int SS = 2048;
static constexpr int DD = 1024;
static constexpr int HH = 16;
static constexpr int DH = 64;

// ---------------------------------------------------------------------------
// WMMA fragment loaders (wave32, 16x16x32 f16).
// A layout (16x32, MxK): lane L(0-15) row M=L holds K {0-7,16-23};
//                        lane L+16  row M=L holds K {8-15,24-31}.
// B layout (32x16, KxN): lane L(0-15) col N=L holds K {0-7,16-23};
//                        lane L+16  col N=L holds K {8-15,24-31}.
// C/D layout (16x16 f32): VGPR r, lanes 0-15 -> (M=r, N=lane);
//                         lanes 16-31 -> (M=r+8, N=lane-16).
// ---------------------------------------------------------------------------
__device__ __forceinline__ v16h load_a16(const _Float16* __restrict__ base, int ld) {
  const int lane = threadIdx.x & 31;
  const _Float16* p = base + (size_t)(lane & 15) * ld + ((lane & 16) ? 8 : 0);
  v8h lo = *(const v8h*)(p);
  v8h hi = *(const v8h*)(p + 16);
  return __builtin_shufflevector(lo, hi, 0,1,2,3,4,5,6,7,8,9,10,11,12,13,14,15);
}

__device__ __forceinline__ v16h load_b16(const _Float16* __restrict__ base, int ld) {
  const int lane = threadIdx.x & 31;
  const _Float16* p = base + (size_t)(lane & 15) * ld + ((lane & 16) ? 8 : 0);
  v8h lo = *(const v8h*)(p);
  v8h hi = *(const v8h*)(p + 16);
  return __builtin_shufflevector(lo, hi, 0,1,2,3,4,5,6,7,8,9,10,11,12,13,14,15);
}

__device__ __forceinline__ v8f wmma_f16(v16h a, v16h b, v8f c) {
  return __builtin_amdgcn_wmma_f32_16x16x32_f16(false, a, false, b, (short)0, c,
                                                false, false);
}

// ---------------------------------------------------------------------------
// fp32 -> f16 conversion (grid-stride)
// ---------------------------------------------------------------------------
__global__ void __launch_bounds__(256) cvt_f32_f16(const float* __restrict__ s,
                                                   _Float16* __restrict__ d, int n) {
  int i = blockIdx.x * blockDim.x + threadIdx.x;
  const int stride = gridDim.x * blockDim.x;
  for (; i < n; i += stride) d[i] = (_Float16)s[i];
}

// ---------------------------------------------------------------------------
// GEMM: out = A[M,K] @ W[N,K]^T   (torch Linear layout)
// Each wave computes a 64x64 tile: 4x4 accumulators, K-step 32 -> 16 WMMA/step.
// MODE 0: store f16 row-major [M,N]            (Q, K)
// MODE 1: store f16 head-transposed [B,H,DH,S] (V)
// MODE 2: store f32 row-major + bias           (final output)
// ---------------------------------------------------------------------------
template <int MODE>
__global__ void __launch_bounds__(256) gemm64(const _Float16* __restrict__ A,
                                              const _Float16* __restrict__ W,
                                              void* __restrict__ out,
                                              const float* __restrict__ bias,
                                              int M, int N, int K) {
  const int lane = threadIdx.x & 31;
  const int wave = threadIdx.x >> 5;
  const int tid  = blockIdx.x * 8 + wave;
  const int ntiles = N >> 6;
  const int m0 = (tid / ntiles) * 64;
  const int n0 = (tid % ntiles) * 64;

  v8f acc[4][4] = {};
  for (int k = 0; k < K; k += 32) {
    v16h a[4], b[4];
#pragma unroll
    for (int i = 0; i < 4; ++i) a[i] = load_a16(A + (size_t)(m0 + 16 * i) * K + k, K);
#pragma unroll
    for (int j = 0; j < 4; ++j) b[j] = load_b16(W + (size_t)(n0 + 16 * j) * K + k, K);
#pragma unroll
    for (int i = 0; i < 4; ++i)
#pragma unroll
      for (int j = 0; j < 4; ++j) acc[i][j] = wmma_f16(a[i], b[j], acc[i][j]);
  }

  const int ro  = (lane & 16) ? 8 : 0;
  const int col = lane & 15;
#pragma unroll
  for (int i = 0; i < 4; ++i)
#pragma unroll
    for (int j = 0; j < 4; ++j)
#pragma unroll
      for (int r = 0; r < 8; ++r) {
        const int m = m0 + 16 * i + r + ro;
        const int n = n0 + 16 * j + col;
        const float v = acc[i][j][r];
        if (MODE == 0) {
          ((_Float16*)out)[(size_t)m * N + n] = (_Float16)v;
        } else if (MODE == 1) {
          const int bb = m >> 11, s = m & (SS - 1);   // S = 2048
          const int h = n >> 6, dh = n & (DH - 1);
          ((_Float16*)out)[((size_t)((bb * HH + h) * DH + dh) << 11) + s] = (_Float16)v;
        } else {
          ((float*)out)[(size_t)m * N + n] = v + bias[n];
        }
      }
}

// ---------------------------------------------------------------------------
// Causal flash attention. One wave per (b, h, 16-query tile).
// Key blocks of 32: 4 WMMA for S = Q K^T (scaled, masked), online softmax,
// P staged through per-wave LDS (C-layout -> A-layout transpose),
// 4 WMMA for O += P @ V  (V pre-transposed [B,H,DH,S]).
// ---------------------------------------------------------------------------
__global__ void __launch_bounds__(256) attn64(const _Float16* __restrict__ Qh,
                                              const _Float16* __restrict__ Kh,
                                              const _Float16* __restrict__ Vt,
                                              _Float16* __restrict__ Zh) {
  const int lane = threadIdx.x & 31;
  const int wave = threadIdx.x >> 5;
  const int gw   = blockIdx.x * 8 + wave;   // 0 .. B*H*(S/16)-1
  const int qt = gw & 127;                  // S/16 = 128
  const int h  = (gw >> 7) & (HH - 1);
  const int b  = gw >> 11;
  const int q0 = qt * 16;

  __shared__ __align__(16) _Float16 pbuf[8][16 * 32];
  _Float16* pb = pbuf[wave];

  const size_t qbase = (size_t)(b * SS + q0) * DD + h * DH;
  const v16h qa0 = load_a16(Qh + qbase, DD);        // dh 0-31
  const v16h qa1 = load_a16(Qh + qbase + 32, DD);   // dh 32-63

  v8f o[4] = {};
  float mrow[8], lrow[8];
#pragma unroll
  for (int r = 0; r < 8; ++r) { mrow[r] = -1e30f; lrow[r] = 0.0f; }

  const int ro  = (lane & 16) ? 8 : 0;
  const int col = lane & 15;
  const float scale = 0.125f;               // 1/sqrt(DH)
  const int nkb = (q0 + 16 + 31) >> 5;      // causal: keys 0 .. q0+15

  for (int kb = 0; kb < nkb; ++kb) {
    const int k0 = kb * 32;
    const size_t kbase = (size_t)(b * SS + k0) * DD + h * DH;

    v8f s_lo = {}, s_hi = {};
    s_lo = wmma_f16(qa0, load_b16(Kh + kbase, DD), s_lo);
    s_lo = wmma_f16(qa1, load_b16(Kh + kbase + 32, DD), s_lo);
    s_hi = wmma_f16(qa0, load_b16(Kh + kbase + (size_t)16 * DD, DD), s_hi);
    s_hi = wmma_f16(qa1, load_b16(Kh + kbase + (size_t)16 * DD + 32, DD), s_hi);

    float plo[8], phi[8], rmax[8];
#pragma unroll
    for (int r = 0; r < 8; ++r) {
      const int qg = q0 + r + ro;
      float a = s_lo[r] * scale;
      float c = s_hi[r] * scale;
      if (k0 + col > qg)      a = -1e30f;   // causal mask
      if (k0 + 16 + col > qg) c = -1e30f;
      plo[r] = a; phi[r] = c;
      rmax[r] = fmaxf(a, c);
    }
#pragma unroll
    for (int msk = 1; msk < 16; msk <<= 1)
#pragma unroll
      for (int r = 0; r < 8; ++r)
        rmax[r] = fmaxf(rmax[r], __shfl_xor(rmax[r], msk, 32));

    float alpha[8], rsum[8];
#pragma unroll
    for (int r = 0; r < 8; ++r) {
      const float mnew = fmaxf(mrow[r], rmax[r]);
      alpha[r] = __expf(mrow[r] - mnew);
      mrow[r]  = mnew;
      plo[r] = __expf(plo[r] - mnew);
      phi[r] = __expf(phi[r] - mnew);
      rsum[r] = plo[r] + phi[r];
    }
#pragma unroll
    for (int msk = 1; msk < 16; msk <<= 1)
#pragma unroll
      for (int r = 0; r < 8; ++r)
        rsum[r] += __shfl_xor(rsum[r], msk, 32);
#pragma unroll
    for (int r = 0; r < 8; ++r) lrow[r] = lrow[r] * alpha[r] + rsum[r];
#pragma unroll
    for (int j = 0; j < 4; ++j)
#pragma unroll
      for (int r = 0; r < 8; ++r) o[j][r] *= alpha[r];

    // C-layout -> A-layout transpose of P through per-wave LDS tile (16x32 f16)
#pragma unroll
    for (int r = 0; r < 8; ++r) {
      const int rw = r + ro;
      pb[rw * 32 + col]      = (_Float16)plo[r];
      pb[rw * 32 + col + 16] = (_Float16)phi[r];
    }
    asm volatile("s_wait_dscnt 0" ::: "memory");   // cross-lane LDS hand-off
    const v16h pa = load_a16(pb, 32);
    asm volatile("s_wait_dscnt 0" ::: "memory");

    const size_t vbase = ((size_t)(b * HH + h) * DH) * SS + k0;
#pragma unroll
    for (int j = 0; j < 4; ++j)
      o[j] = wmma_f16(pa, load_b16(Vt + vbase + (size_t)(16 * j) * SS, SS), o[j]);
  }

  // epilogue: Z[b, s, h*64 + dh] = O / l   (f16, row-major for final GEMM)
#pragma unroll
  for (int j = 0; j < 4; ++j)
#pragma unroll
    for (int r = 0; r < 8; ++r) {
      const int qg = q0 + r + ro;
      const float v = o[j][r] / lrow[r];
      Zh[(size_t)(b * SS + qg) * DD + h * DH + 16 * j + col] = (_Float16)v;
    }
}

// ---------------------------------------------------------------------------
// Launch: cvt(X,Wq,Wk,Wv,Wo) -> gemm(Q) gemm(K) gemm(V^T) -> attn -> gemm(+bias)
// Workspace layout (f16): Xh 16MB | Wq/Wk/Wv/Wo 2MB each | Qh 16MB | Kh 16MB |
//                         Vt 16MB | Zh 16MB   (total 88MB)
// ---------------------------------------------------------------------------
extern "C" void kernel_launch(void* const* d_in, const int* in_sizes, int n_in,
                              void* d_out, int out_size, void* d_ws, size_t ws_size,
                              hipStream_t stream) {
  const float* X  = (const float*)d_in[0];
  const float* Wq = (const float*)d_in[1];
  const float* Wk = (const float*)d_in[2];
  const float* Wv = (const float*)d_in[3];
  const float* Wo = (const float*)d_in[4];
  const float* bo = (const float*)d_in[5];

  const size_t MBy = (size_t)1 << 20;
  char* ws = (char*)d_ws;
  _Float16* Xh  = (_Float16*)(ws + 0 * MBy);
  _Float16* Wqh = (_Float16*)(ws + 16 * MBy);
  _Float16* Wkh = (_Float16*)(ws + 18 * MBy);
  _Float16* Wvh = (_Float16*)(ws + 20 * MBy);
  _Float16* Woh = (_Float16*)(ws + 22 * MBy);
  _Float16* Qh  = (_Float16*)(ws + 24 * MBy);
  _Float16* Kh  = (_Float16*)(ws + 40 * MBy);
  _Float16* Vt  = (_Float16*)(ws + 56 * MBy);
  _Float16* Zh  = (_Float16*)(ws + 72 * MBy);

  const int M = BB * SS, N = DD, K = DD;

  cvt_f32_f16<<<2048, 256, 0, stream>>>(X, Xh, M * K);
  cvt_f32_f16<<<512, 256, 0, stream>>>(Wq, Wqh, N * K);
  cvt_f32_f16<<<512, 256, 0, stream>>>(Wk, Wkh, N * K);
  cvt_f32_f16<<<512, 256, 0, stream>>>(Wv, Wvh, N * K);
  cvt_f32_f16<<<512, 256, 0, stream>>>(Wo, Woh, N * K);

  const int gblocks = (M / 64) * (N / 64) / 8;        // 256 blocks x 8 waves
  gemm64<0><<<gblocks, 256, 0, stream>>>(Xh, Wqh, (void*)Qh, nullptr, M, N, K);
  gemm64<0><<<gblocks, 256, 0, stream>>>(Xh, Wkh, (void*)Kh, nullptr, M, N, K);
  gemm64<1><<<gblocks, 256, 0, stream>>>(Xh, Wvh, (void*)Vt, nullptr, M, N, K);

  attn64<<<BB * HH * (SS / 16) / 8, 256, 0, stream>>>(Qh, Kh, Vt, Zh);

  gemm64<2><<<gblocks, 256, 0, stream>>>(Zh, Woh, d_out, bo, M, N, K);
}
